// Conv4d_25744033973147
// MI455X (gfx1250) — matrix operand order repaired and checked by reading, not compiled, not run on hardware
//
#include <hip/hip_runtime.h>

typedef __bf16 v16bf __attribute__((ext_vector_type(16)));
typedef float  v8f   __attribute__((ext_vector_type(8)));
typedef int    v4i   __attribute__((ext_vector_type(4)));

// problem dims
#define ICN  16
#define OCN  32
#define TTN  8
#define DDN  8
#define HHN  32
#define WWN  32
#define NTAP 81                 // 3*3*3*3 filter taps
#define NWEL (OCN * ICN * NTAP) // 41472 weight elements

// LDS layout (in dwords).  Each dword = packed pair of bf16 (ic even|odd).
//   whi/wlo : [tap 0..81][oc 0..31][icpair 0..7]   (tap 81 = zeros for K-pad)
//   xhi/xlo : [it 0..3][id 0..3][ih 0..3][iw 0..33][icpair 0..7], halos pre-zeroed
#define W_DW   (82 * OCN * 8)        // 20992 dwords per weight array
#define X_DW   (4 * 4 * 4 * 34 * 8)  // 17408 dwords per x array
#define LDS_DW (2 * W_DW + 2 * X_DW) // 76800 dwords = 307,200 B <= 320 KB/WGP

__device__ __forceinline__ unsigned bf16_rne(float f) {
    unsigned u = __float_as_uint(f);
    return (u + 0x7fffu + ((u >> 16) & 1u)) >> 16;   // round-to-nearest-even bf16 bits
}

struct QPair { v4i a, b; };
__device__ __forceinline__ v16bf mk16(v4i a, v4i b) {
    QPair q{a, b};
    return __builtin_bit_cast(v16bf, q);
}

#define WMMA_BF16(ACC, A, B) \
    ACC = __builtin_amdgcn_wmma_f32_16x16x32_bf16(false, A, false, B, (short)0, ACC, false, false)

__global__ __launch_bounds__(256, 1)
void Conv4d_25744033973147_kernel(const float* __restrict__ x,
                                  const float* __restrict__ wgt,
                                  const float* __restrict__ bias,
                                  float* __restrict__ out)
{
    extern __shared__ int smem[];
    int* __restrict__ whi = smem;
    int* __restrict__ wlo = smem + W_DW;
    int* __restrict__ xhi = smem + 2 * W_DW;
    int* __restrict__ xlo = smem + 2 * W_DW + X_DW;

    const int tid = threadIdx.x;

    // block -> (b, t0, d0, h0): covers t in {t0,t0+1}, d in {d0,d0+1}, h in {h0,h0+1}
    const int bid = blockIdx.x;
    const int h0 = (bid & 15) * 2;
    const int d0 = ((bid >> 4) & 3) * 2;
    const int t0 = ((bid >> 6) & 3) * 2;
    const int b  = bid >> 8;

    // ---------------- stage weights: fp32 -> packed bf16 hi/lo, [tap][oc][icpair] ----
    for (int i = tid; i < NWEL; i += 256) {
        int oc  = i / (ICN * NTAP);          // global layout [oc][ic][taps]
        int k   = i - oc * (ICN * NTAP);
        int ic  = k / NTAP;
        int tap = k - ic * NTAP;
        float wv = wgt[i];
        unsigned hb = bf16_rne(wv);
        float    lf = wv - __uint_as_float(hb << 16);   // exact residual
        unsigned lb = bf16_rne(lf);
        int us = ((tap * OCN + oc) * 8 + (ic >> 1)) * 2 + (ic & 1);  // ushort index
        ((unsigned short*)whi)[us] = (unsigned short)hb;
        ((unsigned short*)wlo)[us] = (unsigned short)lb;
    }
    // zero tap 81 (K-padding slot for the odd 9th tap of each (kt,kd) group)
    for (int i = tid; i < OCN * 8; i += 256) {
        whi[NTAP * OCN * 8 + i] = 0;
        wlo[NTAP * OCN * 8 + i] = 0;
    }

    // ---------------- stage x: fp32 -> packed bf16 hi/lo with zeroed halos ----------
    for (int i = tid; i < X_DW; i += 256) {
        int icp = i & 7;
        int j   = i >> 3;           // ((it*4+id)*4+ih)*34 + iw
        int iw  = j % 34;
        int j2  = j / 34;
        int ih  = j2 & 3;
        int id  = (j2 >> 2) & 3;
        int it  = j2 >> 4;
        int t_in = t0 - 1 + it;
        int d_in = d0 - 1 + id;
        int h_in = h0 - 1 + ih;
        int w_in = iw - 1;
        bool ok = ((unsigned)t_in < (unsigned)TTN) & ((unsigned)d_in < (unsigned)DDN) &
                  ((unsigned)h_in < (unsigned)HHN) & ((unsigned)w_in < (unsigned)WWN);
        int tc = t_in < 0 ? 0 : (t_in > 7 ? 7 : t_in);
        int dc = d_in < 0 ? 0 : (d_in > 7 ? 7 : d_in);
        int hc = h_in < 0 ? 0 : (h_in > 31 ? 31 : h_in);
        int wc = w_in < 0 ? 0 : (w_in > 31 ? 31 : w_in);
        int base = ((((b * ICN + 2 * icp) * TTN + tc) * DDN + dc) * HHN + hc) * WWN + wc;
        float v0 = x[base];
        float v1 = x[base + TTN * DDN * HHN * WWN];   // +1 input channel
        v0 = ok ? v0 : 0.0f;
        v1 = ok ? v1 : 0.0f;
        unsigned h0b = bf16_rne(v0), h1b = bf16_rne(v1);
        float l0 = v0 - __uint_as_float(h0b << 16);
        float l1 = v1 - __uint_as_float(h1b << 16);
        xhi[i] = (int)(h0b | (h1b << 16));
        xlo[i] = (int)(bf16_rne(l0) | (bf16_rne(l1) << 16));
    }
    __syncthreads();

    // ---------------- per-wave tile: (t,d,h) x full 32-w row x 32 oc -----------------
    const int lane = tid & 31;
    const int wv   = tid >> 5;              // 0..7 = 2t x 2d x 2h
    const int hs   = wv & 1;
    const int t    = t0 + (wv >> 2);
    const int d    = d0 + ((wv >> 1) & 1);
    const int h    = h0 + hs;

    const int col = lane & 15;              // M(oc) for A, N(w) for B/D
    const int hif = lane >> 4;              // K half select
    const int aLane = col * 32;             // bytes inside one 1024B weight tap block
    const char* whiB = (const char*)whi;
    const char* wloB = (const char*)wlo;
    const char* xhiB = (const char*)xhi;
    const char* xloB = (const char*)xlo;

    v8f acc00 = {};   // oc  0..15, w  0..15
    v8f acc01 = {};   // oc  0..15, w 16..31
    v8f acc10 = {};   // oc 16..31, w  0..15
    v8f acc11 = {};   // oc 16..31, w 16..31

    for (int kt = 0; kt < 3; ++kt) {
        const int t_in = t + kt - 1;
        if ((unsigned)t_in >= (unsigned)TTN) continue;       // wave-uniform skip
        const int it = t - t0 + kt;
        for (int kd = 0; kd < 3; ++kd) {
            const int d_in = d + kd - 1;
            if ((unsigned)d_in >= (unsigned)DDN) continue;   // wave-uniform skip
            const int id    = d - d0 + kd;
            const int base9 = (kt * 3 + kd) * 9;
            const int tdOff = (it * 4 + id) * 4;
            #pragma unroll
            for (int p = 0; p < 5; ++p) {                    // 9 taps + 1 pad, pairs
                const int tl0  = 2 * p;
                const int tl1  = 2 * p + 1;
                const int tl1c = (tl1 > 8) ? 8 : tl1;        // B pad -> clamp (finite)
                const int kh0 = tl0 / 3, kw0 = tl0 % 3;
                const int kh1 = tl1c / 3, kw1 = tl1c % 3;
                const int tapA0 = base9 + tl0;
                const int tapA1 = (tl1 <= 8) ? (base9 + tl1) : NTAP;  // A pad -> zeros

                // B: x tile, lanes 0-15 = tap0, lanes 16-31 = tap1; n-tile1 = +512B
                int offB0 = ((tdOff + hs + kh0) * 34 + kw0) * 32;
                int offB1 = ((tdOff + hs + kh1) * 34 + kw1) * 32;
                int offB  = (hif ? offB1 : offB0) + col * 32;
                v4i bh0a = *(const v4i*)(xhiB + offB);
                v4i bh1a = *(const v4i*)(xhiB + offB + 16);
                v4i bl0a = *(const v4i*)(xloB + offB);
                v4i bl1a = *(const v4i*)(xloB + offB + 16);
                v4i bh0b = *(const v4i*)(xhiB + offB + 512);
                v4i bh1b = *(const v4i*)(xhiB + offB + 528);
                v4i bl0b = *(const v4i*)(xloB + offB + 512);
                v4i bl1b = *(const v4i*)(xloB + offB + 528);

                // A: weights, lanes 0-15 = tap0, lanes 16-31 = tap1; oc-tile1 = +512B
                int offA0 = (hif ? tapA1 : tapA0) * 1024 + aLane;
                int offA1 = offA0 + 16;
                v4i ah00 = *(const v4i*)(whiB + offA0);
                v4i ah01 = *(const v4i*)(whiB + offA1);
                v4i al00 = *(const v4i*)(wloB + offA0);
                v4i al01 = *(const v4i*)(wloB + offA1);
                v4i ah10 = *(const v4i*)(whiB + offA0 + 512);
                v4i ah11 = *(const v4i*)(whiB + offA1 + 512);
                v4i al10 = *(const v4i*)(wloB + offA0 + 512);
                v4i al11 = *(const v4i*)(wloB + offA1 + 512);

                v16bf BhiA = mk16(bh0a, bh1a);
                v16bf BloA = mk16(bl0a, bl1a);
                v16bf BhiB = mk16(bh0b, bh1b);
                v16bf BloB = mk16(bl0b, bl1b);
                v16bf Ahi0 = mk16(ah00, ah01);
                v16bf Alo0 = mk16(al00, al01);
                v16bf Ahi1 = mk16(ah10, ah11);
                v16bf Alo1 = mk16(al10, al11);

                // fp32 emulation: hi*hi + hi*lo + lo*hi (lo*lo dropped, ~2^-18)
                // 4 independent accumulator chains, interleaved for pipelining
                WMMA_BF16(acc00, Ahi0, BhiA);
                WMMA_BF16(acc01, Ahi0, BhiB);
                WMMA_BF16(acc10, Ahi1, BhiA);
                WMMA_BF16(acc11, Ahi1, BhiB);
                WMMA_BF16(acc00, Ahi0, BloA);
                WMMA_BF16(acc01, Ahi0, BloB);
                WMMA_BF16(acc10, Ahi1, BloA);
                WMMA_BF16(acc11, Ahi1, BloB);
                WMMA_BF16(acc00, Alo0, BhiA);
                WMMA_BF16(acc01, Alo0, BhiB);
                WMMA_BF16(acc10, Alo1, BhiA);
                WMMA_BF16(acc11, Alo1, BhiB);
            }
        }
    }

    // ---------------- store: D vgpr v -> oc = v + 8*hif (+16 for acc1x), N = col -----
    const int spatial  = ((t * DDN + d) * HHN + h) * WWN + col;
    const int ocStride = TTN * DDN * HHN * WWN;   // 65536
    #pragma unroll
    for (int v = 0; v < 8; ++v) {
        const int oc0 = v + 8 * hif;
        const int oc1 = oc0 + 16;
        float b0 = bias[oc0], b1 = bias[oc1];
        out[(b * OCN + oc0) * ocStride + spatial]      = acc00[v] + b0;
        out[(b * OCN + oc0) * ocStride + spatial + 16] = acc01[v] + b0;
        out[(b * OCN + oc1) * ocStride + spatial]      = acc10[v] + b1;
        out[(b * OCN + oc1) * ocStride + spatial + 16] = acc11[v] + b1;
    }
}

extern "C" void kernel_launch(void* const* d_in, const int* in_sizes, int n_in,
                              void* d_out, int out_size, void* d_ws, size_t ws_size,
                              hipStream_t stream) {
    const float* x    = (const float*)d_in[0];
    const float* wgt  = (const float*)d_in[1];
    const float* bias = (const float*)d_in[2];
    float* out = (float*)d_out;

    const int blocks = 2 * 4 * 4 * 16;                 // (b) x (t/2) x (d/2) x (h/2)
    const size_t ldsBytes = (size_t)LDS_DW * sizeof(int);  // 307,200 B
    Conv4d_25744033973147_kernel<<<blocks, 256, ldsBytes, stream>>>(x, wgt, bias, out);
}
